// NeuralProgramInterpreter_72997264163011
// MI455X (gfx1250) — matrix operand order repaired and checked by reading, not compiled, not run on hardware
//
#include <hip/hip_runtime.h>

// ---------------------------------------------------------------------------
// NeuralProgramInterpreter on MI455X (gfx1250, wave32, WMMA bf16 16x16x32)
//
// 16 sequential steps of
//   transformed[b,k] = sum_o sum_d w[b,o] * h[b,d] * K[o,d,k]
//   h = sigmoid(gate)*transformed + (1-sigmoid(gate))*h
// B=64, S=16, D=1024, O=31.
//
// v2: A matrix (w-scaled h, bf16) is materialized per step in WMMA A-fragment
// layout, so the GEMM inner loop is pure {load A-frag, load 2 B-frags,
// 2x WMMA} with zero per-iteration mul/cvt VALU. K is converted once to bf16
// pre-swizzled into B-fragment layout (65 MB, L2-resident on 192 MB L2).
// ---------------------------------------------------------------------------

typedef __attribute__((ext_vector_type(16))) __bf16 v16bf;
typedef __attribute__((ext_vector_type(8)))  float  v8f;

#define B_ 64
#define S_ 16
#define O_ 31
#define D_ 1024
#define NSPLIT 8

// workspace byte offsets (16B aligned)
#define OFF_KBF    0ull          // 31*1024*1024 bf16 = 65,011,712 B
#define OFF_PART   65011712ull   // 8*64*1024 fp32    =  2,097,152 B
#define OFF_APACK  67108864ull   // 31*32*4*32*16 bf16=  4,063,232 B
#define OFF_WSM    71172096ull   // 64*16*31 fp32     =    126,976 B
#define OFF_GATE   71299072ull   // 64*16 fp32        =      4,096 B
#define OFF_H      71303168ull   // 64*1024 fp32      =    262,144 B
                                 // total ~68.3 MB

// ---------------------------------------------------------------------------
// Pack op_kernels (O,D,D) fp32 -> bf16 in WMMA B-fragment order:
// frag index = ((o*32 + kb)*64 + nb)*32 + lane ; 16 bf16 per lane:
//   elem j -> K[o][ kb*32 + ((lane&16)?8:0) + (j&7) + ((j&8)?16:0) ][ nb*16 + (lane&15) ]
// ---------------------------------------------------------------------------
__global__ void __launch_bounds__(256)
pack_kernel(const float* __restrict__ K, __bf16* __restrict__ Kbf) {
    unsigned idx  = blockIdx.x * 256u + threadIdx.x;   // < 31 * 1M
    unsigned j    = idx & 15u;
    unsigned lane = (idx >> 4) & 31u;
    unsigned nb   = (idx >> 9) & 63u;
    unsigned kb   = (idx >> 15) & 31u;
    unsigned o    = idx >> 20;
    if (o >= O_) return;
    unsigned d = kb * 32u + ((lane & 16u) ? 8u : 0u) + (j & 7u) + ((j & 8u) ? 16u : 0u);   // reduction dim
    unsigned n = nb * 16u + (lane & 15u);                    // output dim
    Kbf[idx] = (__bf16)K[((size_t)o * D_ + d) * D_ + n];
}

// ---------------------------------------------------------------------------
// Softmax over opcode logits (per (b,s) row of 31) + sigmoid gates.
// ---------------------------------------------------------------------------
__global__ void __launch_bounds__(256)
softgate_kernel(const float* __restrict__ logits, const float* __restrict__ operands,
                float* __restrict__ wsm, float* __restrict__ gates) {
    int t = blockIdx.x * 256 + threadIdx.x;          // (b*16 + s)
    if (t >= B_ * S_) return;
    const float* row = logits + (size_t)t * O_;
    float m = row[0];
    for (int o = 1; o < O_; ++o) m = fmaxf(m, row[o]);
    float e[O_], s = 0.f;
    for (int o = 0; o < O_; ++o) { e[o] = __expf(row[o] - m); s += e[o]; }
    float inv = 1.f / s;
    for (int o = 0; o < O_; ++o) wsm[(size_t)t * O_ + o] = e[o] * inv;
    float gl = operands[(size_t)t * 4 + 3];
    gates[t] = 1.f / (1.f + __expf(-gl));
}

// ---------------------------------------------------------------------------
// Per-step A pack: U[b, o*D+d] = w[b,o]*h[b,d] (bf16), stored in WMMA
// A-fragment (16x32 bf16) layout. One thread builds one 16-elem fragment:
// frag index = ((o*32 + kb)*4 + mt)*32 + lane ; per lane m = mt*16+(lane&15),
// kbase=(lane&16)?8:0; elems 0..7 -> d=kb*32+kbase+j, 8..15 -> +16.
// ---------------------------------------------------------------------------
__global__ void __launch_bounds__(256)
aprep_kernel(const float* __restrict__ hprev, const float* __restrict__ wsm,
             v16bf* __restrict__ Apack, int step) {
    unsigned idx  = blockIdx.x * 256u + threadIdx.x;   // < 31*32*4*32 = 126976
    unsigned lane = idx & 31u;
    unsigned mt   = (idx >> 5) & 3u;
    unsigned kb   = (idx >> 7) & 31u;
    unsigned o    = idx >> 12;
    if (o >= O_) return;
    const unsigned m     = mt * 16u + (lane & 15u);
    const unsigned dbase = kb * 32u + ((lane & 16u) ? 8u : 0u);
    const float scale = wsm[((size_t)m * S_ + step) * O_ + o];
    const float* __restrict__ hrow = hprev + (size_t)m * D_;
    float4 x0 = *(const float4*)(hrow + dbase);
    float4 x1 = *(const float4*)(hrow + dbase + 4);
    float4 x2 = *(const float4*)(hrow + dbase + 16);
    float4 x3 = *(const float4*)(hrow + dbase + 20);
    v16bf a;
    a[0]  = (__bf16)(x0.x * scale); a[1]  = (__bf16)(x0.y * scale);
    a[2]  = (__bf16)(x0.z * scale); a[3]  = (__bf16)(x0.w * scale);
    a[4]  = (__bf16)(x1.x * scale); a[5]  = (__bf16)(x1.y * scale);
    a[6]  = (__bf16)(x1.z * scale); a[7]  = (__bf16)(x1.w * scale);
    a[8]  = (__bf16)(x2.x * scale); a[9]  = (__bf16)(x2.y * scale);
    a[10] = (__bf16)(x2.z * scale); a[11] = (__bf16)(x2.w * scale);
    a[12] = (__bf16)(x3.x * scale); a[13] = (__bf16)(x3.y * scale);
    a[14] = (__bf16)(x3.z * scale); a[15] = (__bf16)(x3.w * scale);
    Apack[idx] = a;
}

// ---------------------------------------------------------------------------
// One scan step GEMM. 1024 waves: (mt 0..3) x (ntp 0..31) x (slice 0..7);
// each wave computes a 16x32 C region (two tiles sharing each A fragment).
// Inner loop: 1 A-frag load + 2 B-frag loads + 2 WMMAs. Split-K by opcode.
// ---------------------------------------------------------------------------
__global__ void __launch_bounds__(256)
gemm_kernel(const v16bf* __restrict__ Apack, const v16bf* __restrict__ Bpack,
            float* __restrict__ partial) {
    const int lane  = threadIdx.x & 31;
    const int warp  = threadIdx.x >> 5;
    const int gw    = blockIdx.x * 8 + warp;   // 0..1023
    const int slice = gw & (NSPLIT - 1);
    const int ntp   = (gw >> 3) & 31;
    const int mt    = gw >> 8;                 // 0..3
    const int b0    = mt * 16, n0 = ntp * 32;
    const int nt0   = ntp * 2;

    v8f acc0 = {};
    v8f acc1 = {};
    for (int o = slice; o < O_; o += NSPLIT) {
        // A frags: ((o*32+kb)*4 + mt)*32 + lane ; kb stride = 128
        const v16bf* __restrict__ Ap = Apack + ((size_t)(o * 128 + mt)) * 32 + lane;
        // B frags: ((o*32+kb)*64 + nt)*32 + lane ; kb stride = 2048, nt stride = 32
        const v16bf* __restrict__ Bp = Bpack + ((size_t)(o * 2048 + nt0)) * 32 + lane;
#pragma unroll 4
        for (int kb = 0; kb < 32; ++kb) {
            v16bf a  = Ap[(size_t)kb * 128];
            v16bf f0 = Bp[(size_t)kb * 2048];
            v16bf f1 = Bp[(size_t)kb * 2048 + 32];
            acc0 = __builtin_amdgcn_wmma_f32_16x16x32_bf16(
                false, a, false, f0, (short)0, acc0, false, false);
            acc1 = __builtin_amdgcn_wmma_f32_16x16x32_bf16(
                false, a, false, f1, (short)0, acc1, false, false);
        }
    }
    // C layout: VGPR r -> M = r + ((lane&16)?8:0), N = lane&15
    const int mbase = b0 + ((lane & 16) ? 8 : 0);
    float* __restrict__ pout =
        partial + (size_t)slice * (B_ * D_) + (size_t)n0 + (lane & 15);
#pragma unroll
    for (int r = 0; r < 8; ++r) {
        pout[(size_t)(mbase + r) * D_]      = acc0[r];
        pout[(size_t)(mbase + r) * D_ + 16] = acc1[r];
    }
}

// ---------------------------------------------------------------------------
// Sum 8 split-K partials, apply sigmoid-gate blend. Deterministic.
// ---------------------------------------------------------------------------
__global__ void __launch_bounds__(256)
reduce_kernel(const float* __restrict__ hprev, const float* __restrict__ partial,
              const float* __restrict__ gates, float* __restrict__ hout, int step) {
    int t = blockIdx.x * 256 + threadIdx.x;    // < 64*1024
    float s = 0.f;
#pragma unroll
    for (int p = 0; p < NSPLIT; ++p) s += partial[(size_t)p * (B_ * D_) + t];
    int b = t >> 10;
    float g = gates[b * S_ + step];
    hout[t] = g * s + (1.f - g) * hprev[t];
}

// ---------------------------------------------------------------------------
extern "C" void kernel_launch(void* const* d_in, const int* in_sizes, int n_in,
                              void* d_out, int out_size, void* d_ws, size_t ws_size,
                              hipStream_t stream) {
    (void)in_sizes; (void)n_in; (void)out_size; (void)ws_size;
    const float* logits   = (const float*)d_in[0];   // (64,16,31)
    const float* operands = (const float*)d_in[1];   // (64,16,4)
    const float* signal   = (const float*)d_in[2];   // (64,1024)
    const float* opk      = (const float*)d_in[3];   // (31,1024,1024)
    float* out = (float*)d_out;                      // (64,1024)

    char* ws = (char*)d_ws;
    __bf16* kbf    = (__bf16*)(ws + OFF_KBF);
    float*  partial = (float*)(ws + OFF_PART);
    v16bf*  apack   = (v16bf*)(ws + OFF_APACK);
    float*  wsm     = (float*)(ws + OFF_WSM);
    float*  gates   = (float*)(ws + OFF_GATE);
    float*  hws     = (float*)(ws + OFF_H);

    pack_kernel<<<(O_ * 1024 * 1024) / 256, 256, 0, stream>>>(opk, kbf);
    softgate_kernel<<<(B_ * S_ + 255) / 256, 256, 0, stream>>>(logits, operands, wsm, gates);

    // h ping-pong: step s reads hp, writes ho; step 15 writes d_out.
    for (int s = 0; s < S_; ++s) {
        const float* hp = (s == 0) ? signal : ((s & 1) ? hws : out);
        float* ho = (s & 1) ? out : hws;
        aprep_kernel<<<496, 256, 0, stream>>>(hp, wsm, apack, s);
        gemm_kernel<<<128, 256, 0, stream>>>(apack, (const v16bf*)kbf, partial);
        reduce_kernel<<<(B_ * D_) / 256, 256, 0, stream>>>(hp, partial, gates, ho, s);
    }
}
